// MambaBlock_16990890623505
// MI455X (gfx1250) — matrix operand order repaired and checked by reading, not compiled, not run on hardware
//
#include <hip/hip_runtime.h>
#include <hip/hip_bf16.h>

// ---------------- types ----------------
typedef __attribute__((ext_vector_type(16))) __bf16 bf16x16;
typedef __attribute__((ext_vector_type(8)))  __bf16 bf16x8;
typedef __attribute__((ext_vector_type(8)))  float  f32x8;

#define D_MODEL 2048
#define D_INNER 4096
#define SEQ     2048
#define DT_RANK 128
#define D_STATE 16
#define NPAD    256   // DT_RANK + 2*D_STATE = 160 padded to 256

__device__ __forceinline__ unsigned short f32_to_bf16_bits(float f) {
    unsigned int u = __float_as_uint(f);
    unsigned int r = 0x7FFFu + ((u >> 16) & 1u);
    u += r;
    return (unsigned short)(u >> 16);
}

__device__ __forceinline__ float sigmoidf_(float x) { return 1.0f / (1.0f + __expf(-x)); }
__device__ __forceinline__ float siluf_(float x)    { return x * sigmoidf_(x); }
__device__ __forceinline__ float softplusf_(float x){ return (x > 20.0f) ? x : logf(1.0f + __expf(x)); }

// ---------------- f32 -> bf16 conversion with optional row padding ----------------
__global__ __launch_bounds__(256) void cvt_bf16_kernel(const float* __restrict__ src,
                                                       unsigned short* __restrict__ dst,
                                                       int src_rows, int dst_rows, int cols) {
    size_t i = (size_t)blockIdx.x * 256 + threadIdx.x;
    size_t total = (size_t)dst_rows * cols;
    if (i >= total) return;
    int r = (int)(i / cols);
    int c = (int)(i % cols);
    dst[i] = (r < src_rows) ? f32_to_bf16_bits(src[(size_t)r * cols + c]) : (unsigned short)0;
}

// ---------------- WMMA bf16 GEMM:  C[m,n] = sum_k A[m,k] * W[n,k]  (NT) ----------------
// Block = 8 waves (2 x 4), block tile 128(M) x 256(N); wave tile 64 x 64 = 4x4 WMMA tiles.
// M % 128 == 0, N % 256 == 0, K % 32 == 0 guaranteed by caller (padding).
// MODE 0: plain store.  MODE 1: C = softplus(acc + bias[col]).
template <int MODE>
__global__ __launch_bounds__(256) void gemm_wmma_bf16(const __bf16* __restrict__ A,
                                                      const __bf16* __restrict__ W,
                                                      float* __restrict__ C,
                                                      const float* __restrict__ bias,
                                                      int M, int N, int K, int ldc) {
    const int lane = threadIdx.x & 31;
    const int wave = threadIdx.x >> 5;
    const int wm = wave >> 2;         // 0..1
    const int wn = wave & 3;          // 0..3
    const int m0 = blockIdx.y * 128 + wm * 64;
    const int n0 = blockIdx.x * 256 + wn * 64;
    const int half = lane >> 4;       // 0/1
    const int l16  = lane & 15;

    f32x8 acc[4][4] = {};

    for (int k0 = 0; k0 < K; k0 += 32) {
        bf16x16 af[4];
        bf16x16 bfv[4];
        // A fragments: lanes 0-15 -> K 0..15 ; lanes 16-31 -> K 16..31
#pragma unroll
        for (int i = 0; i < 4; ++i) {
            const __bf16* ap = A + (size_t)(m0 + 16 * i + l16) * K + (k0 + 16 * half);
            bf16x8 c0 = *(const bf16x8*)(ap);
            bf16x8 c1 = *(const bf16x8*)(ap + 8);
            bf16x16 f;
#pragma unroll
            for (int j = 0; j < 8; ++j) { f[j] = c0[j]; f[j + 8] = c1[j]; }
            af[i] = f;
        }
        // B fragments: lanes 0-15 -> K 0..15 ; lanes 16-31 -> K 16..31 (contiguous 32B)
#pragma unroll
        for (int j = 0; j < 4; ++j) {
            const __bf16* bp = W + (size_t)(n0 + 16 * j + l16) * K + (k0 + 16 * half);
            bfv[j] = *(const bf16x16*)(bp);
        }
#pragma unroll
        for (int i = 0; i < 4; ++i)
#pragma unroll
            for (int j = 0; j < 4; ++j)
                acc[i][j] = __builtin_amdgcn_wmma_f32_16x16x32_bf16(
                    false, af[i], false, bfv[j], (short)0, acc[i][j], false, false);
    }

    // C/D layout: element r of v8f -> row = 16*i + 8*half + r, col = 16*j + l16
#pragma unroll
    for (int i = 0; i < 4; ++i) {
#pragma unroll
        for (int j = 0; j < 4; ++j) {
            const int col = n0 + 16 * j + l16;
#pragma unroll
            for (int r = 0; r < 8; ++r) {
                const int row = m0 + 16 * i + 8 * half + r;
                float v = acc[i][j][r];
                if (MODE == 1) v = softplusf_(v + bias[col]);
                C[(size_t)row * ldc + col] = v;
            }
        }
    }
}

// ---------------- depthwise causal conv (K=4) + bias + SiLU, plus silu(z) ----------------
// in : proj layout (e, s), e in [0, 2*D_INNER)
// out: h  f32 (s, d)  [conv+silu of rows 0..D_INNER)
//      h  bf16 (s, d) for the W_x GEMM
//      silu(z) f32 (s, d)  [rows D_INNER..2*D_INNER)
__global__ __launch_bounds__(256) void conv_silu_kernel(const float* __restrict__ proj,
                                                        const float* __restrict__ cw,
                                                        const float* __restrict__ cb,
                                                        float* __restrict__ hsd,
                                                        unsigned short* __restrict__ hbf_t,
                                                        float* __restrict__ zsil) {
    int idx = blockIdx.x * 256 + threadIdx.x;            // 2*D_INNER*SEQ threads
    int d = idx >> 11;                                   // /SEQ
    int s = idx & (SEQ - 1);
    const float* row = proj + (size_t)d * SEQ;
    if (d < D_INNER) {
        float acc = cb[d];
#pragma unroll
        for (int j = 0; j < 4; ++j) {
            int sp = s - 3 + j;
            if (sp >= 0) acc += cw[d * 4 + j] * row[sp];
        }
        float v = siluf_(acc);
        hsd[(size_t)s * D_INNER + d]   = v;
        hbf_t[(size_t)s * D_INNER + d] = f32_to_bf16_bits(v);
    } else {
        int dz = d - D_INNER;
        zsil[(size_t)s * D_INNER + dz] = siluf_(row[s]);
    }
}

// ---------------- rmsnorm of delta / B / C slices of ssm_p (padded ld = NPAD) ----------------
__global__ __launch_bounds__(256) void rmsnorm_dbc_kernel(const float* __restrict__ ssmp,
                                                          const float* __restrict__ dt_ln_w,
                                                          const float* __restrict__ B_ln_w,
                                                          const float* __restrict__ C_ln_w,
                                                          unsigned short* __restrict__ delta_bf,
                                                          float* __restrict__ Bm,
                                                          float* __restrict__ Cm) {
    int s = blockIdx.x;
    int t = threadIdx.x;
    __shared__ float sq[160];
    __shared__ float inv0, inv1, inv2;
    float v = 0.0f;
    if (t < 160) { v = ssmp[(size_t)s * NPAD + t]; sq[t] = v * v; }
    __syncthreads();
    if (t == 0) {
        float a = 0.f; for (int i = 0;   i < 128; ++i) a += sq[i];
        float b = 0.f; for (int i = 128; i < 144; ++i) b += sq[i];
        float c = 0.f; for (int i = 144; i < 160; ++i) c += sq[i];
        inv0 = rsqrtf(a / 128.0f + 1e-6f);
        inv1 = rsqrtf(b / 16.0f  + 1e-6f);
        inv2 = rsqrtf(c / 16.0f  + 1e-6f);
    }
    __syncthreads();
    if (t < 128) {
        delta_bf[(size_t)s * DT_RANK + t] = f32_to_bf16_bits(dt_ln_w[t] * v * inv0);
    } else if (t < 144) {
        int n = t - 128;
        float r = B_ln_w[n] * v * inv1;
        r = (float)(_Float16)r;            // reference's fp16 round-trip
        Bm[(size_t)s * D_STATE + n] = r;
    } else if (t < 160) {
        int n = t - 144;
        float r = C_ln_w[n] * v * inv2;
        r = (float)(_Float16)r;
        Cm[(size_t)s * D_STATE + n] = r;
    }
}

// ---------------- sequential selective scan (one thread per channel) ----------------
// All per-step streams are (s, d): lane-contiguous, fully coalesced per time-step.
#define TCH 128
__global__ __launch_bounds__(256) void scan_kernel(const float* __restrict__ dt,   // (s,d)
                                                   const float* __restrict__ hsd,  // (s,d)
                                                   const float* __restrict__ zsil, // (s,d) silu(z)
                                                   const float* __restrict__ Bm,   // (s,16)
                                                   const float* __restrict__ Cm,   // (s,16)
                                                   const float* __restrict__ A_log,// (d,16)
                                                   const float* __restrict__ Dp,   // (d)
                                                   float* __restrict__ yraw) {     // (s,d)
    const int d = blockIdx.x * 256 + threadIdx.x;
    float Ad[D_STATE], st[D_STATE];
#pragma unroll
    for (int n = 0; n < D_STATE; ++n) { Ad[n] = -__expf(A_log[(size_t)d * D_STATE + n]); st[n] = 0.0f; }
    const float Dpd = Dp[d];
    __shared__ float sB[TCH * D_STATE];
    __shared__ float sC[TCH * D_STATE];

    for (int tc = 0; tc < SEQ; tc += TCH) {
        for (int i = threadIdx.x; i < TCH * D_STATE; i += 256) {
            sB[i] = Bm[(size_t)tc * D_STATE + i];
            sC[i] = Cm[(size_t)tc * D_STATE + i];
        }
        __syncthreads();
        for (int tt = 0; tt < TCH; ++tt) {
            const int t = tc + tt;
            const size_t ofs = (size_t)t * D_INNER + d;
            const float dtv = dt[ofs];
            const float hv  = hsd[ofs];
            const float zs  = zsil[ofs];
            // overlap next-chunk latency with the serial recurrence (speculative prefetch,
            // silently dropped past end-of-buffer per ISA)
            const size_t pofs = ofs + (size_t)TCH * D_INNER;
            __builtin_prefetch(&dt[pofs],   0, 0);
            __builtin_prefetch(&hsd[pofs],  0, 0);
            __builtin_prefetch(&zsil[pofs], 0, 0);
            float yt = 0.0f;
#pragma unroll
            for (int n = 0; n < D_STATE; ++n) {
                st[n] = __expf(dtv * Ad[n]) * st[n] + dtv * sB[tt * D_STATE + n] * hv;
                yt += st[n] * sC[tt * D_STATE + n];
            }
            float yv = yt + Dpd * hv;
            yv *= zs;
            yraw[ofs] = yv;
        }
        __syncthreads();
    }
}

// ---------------- final rmsnorm over d (4096) -> bf16 (s,d) ----------------
__global__ __launch_bounds__(256) void rmsnorm_scan_kernel(const float* __restrict__ yraw,
                                                           const float* __restrict__ w,
                                                           unsigned short* __restrict__ ybf) {
    const int s = blockIdx.x;
    const float* row = yraw + (size_t)s * D_INNER;
    float ss = 0.0f;
    for (int d2 = threadIdx.x; d2 < D_INNER; d2 += 256) { float v = row[d2]; ss += v * v; }
    __shared__ float red[256];
    red[threadIdx.x] = ss;
    __syncthreads();
    for (int off = 128; off > 0; off >>= 1) {
        if ((int)threadIdx.x < off) red[threadIdx.x] += red[threadIdx.x + off];
        __syncthreads();
    }
    const float inv = rsqrtf(red[0] / (float)D_INNER + 1e-6f);
    for (int d2 = threadIdx.x; d2 < D_INNER; d2 += 256)
        ybf[(size_t)s * D_INNER + d2] = f32_to_bf16_bits(w[d2] * row[d2] * inv);
}

// ---------------- launcher ----------------
extern "C" void kernel_launch(void* const* d_in, const int* in_sizes, int n_in,
                              void* d_out, int out_size, void* d_ws, size_t ws_size,
                              hipStream_t stream) {
    (void)in_sizes; (void)n_in; (void)out_size; (void)ws_size;
    const float* x        = (const float*)d_in[0];
    const float* W_in     = (const float*)d_in[2];
    const float* conv_w   = (const float*)d_in[3];
    const float* conv_b   = (const float*)d_in[4];
    const float* W_x      = (const float*)d_in[5];
    const float* W_dt     = (const float*)d_in[6];
    const float* b_dt     = (const float*)d_in[7];
    const float* A_log    = (const float*)d_in[8];
    const float* Dp       = (const float*)d_in[9];
    const float* dt_ln_w  = (const float*)d_in[10];
    const float* B_ln_w   = (const float*)d_in[11];
    const float* C_ln_w   = (const float*)d_in[12];
    const float* scan_ln_w= (const float*)d_in[13];
    const float* W_out    = (const float*)d_in[14];
    float* out            = (float*)d_out;

    size_t off = 0;
    char* base = (char*)d_ws;
    auto alloc = [&](size_t bytes) -> char* {
        char* p = base + off;
        off += (bytes + 255) & ~(size_t)255;
        return p;
    };
    unsigned short* xbf   = (unsigned short*)alloc((size_t)SEQ * D_MODEL * 2);
    unsigned short* wibf  = (unsigned short*)alloc((size_t)2 * D_INNER * D_MODEL * 2);
    float*          proj  = (float*)        alloc((size_t)2 * D_INNER * SEQ * 4);
    float*          hsd   = (float*)        alloc((size_t)SEQ * D_INNER * 4);
    unsigned short* hbft  = (unsigned short*)alloc((size_t)SEQ * D_INNER * 2);
    float*          zsil  = (float*)        alloc((size_t)SEQ * D_INNER * 4);
    unsigned short* wxbf  = (unsigned short*)alloc((size_t)NPAD * D_INNER * 2);
    float*          ssmp  = (float*)        alloc((size_t)SEQ * NPAD * 4);
    unsigned short* dbf   = (unsigned short*)alloc((size_t)SEQ * DT_RANK * 2);
    float*          Bm    = (float*)        alloc((size_t)SEQ * D_STATE * 4);
    float*          Cm    = (float*)        alloc((size_t)SEQ * D_STATE * 4);
    unsigned short* wdtbf = (unsigned short*)alloc((size_t)D_INNER * DT_RANK * 2);
    float*          dtm   = (float*)        alloc((size_t)SEQ * D_INNER * 4);
    float*          yraw  = (float*)        alloc((size_t)SEQ * D_INNER * 4);
    unsigned short* ybf   = (unsigned short*)alloc((size_t)SEQ * D_INNER * 2);
    unsigned short* wobf  = (unsigned short*)alloc((size_t)D_MODEL * D_INNER * 2);

    auto cvt = [&](const float* src, unsigned short* dst, int sr, int dr, int cols) {
        size_t total = (size_t)dr * cols;
        cvt_bf16_kernel<<<(unsigned)((total + 255) / 256), 256, 0, stream>>>(src, dst, sr, dr, cols);
    };
    cvt(x,     xbf,   SEQ,          SEQ,          D_MODEL);
    cvt(W_in,  wibf,  2 * D_INNER,  2 * D_INNER,  D_MODEL);
    cvt(W_x,   wxbf,  DT_RANK + 2 * D_STATE, NPAD, D_INNER);
    cvt(W_dt,  wdtbf, D_INNER,      D_INNER,      DT_RANK);
    cvt(W_out, wobf,  D_MODEL,      D_MODEL,      D_INNER);

    // GEMM1: proj[e,s] = sum_d W_in[e,d] * x[s,d]   (M=8192,N=2048,K=2048)
    gemm_wmma_bf16<0><<<dim3(SEQ / 256, (2 * D_INNER) / 128), 256, 0, stream>>>(
        (const __bf16*)wibf, (const __bf16*)xbf, proj, nullptr,
        2 * D_INNER, SEQ, D_MODEL, SEQ);

    // conv + SiLU on h; silu(z); both emitted in (s,d) layout
    conv_silu_kernel<<<(2 * D_INNER * SEQ) / 256, 256, 0, stream>>>(
        proj, conv_w, conv_b, hsd, hbft, zsil);

    // GEMM2: ssm_p[s,e'] = sum_d h[s,d] * W_x[e',d]  (M=2048,N=256pad,K=4096)
    gemm_wmma_bf16<0><<<dim3(NPAD / 256, SEQ / 128), 256, 0, stream>>>(
        (const __bf16*)hbft, (const __bf16*)wxbf, ssmp, nullptr,
        SEQ, NPAD, D_INNER, NPAD);

    // rmsnorm delta / B / C
    rmsnorm_dbc_kernel<<<SEQ, 256, 0, stream>>>(ssmp, dt_ln_w, B_ln_w, C_ln_w, dbf, Bm, Cm);

    // GEMM3: dt[s,d] = softplus(sum_r delta[s,r]*W_dt[d,r] + b_dt[d])  (M=2048,N=4096,K=128)
    gemm_wmma_bf16<1><<<dim3(D_INNER / 256, SEQ / 128), 256, 0, stream>>>(
        (const __bf16*)dbf, (const __bf16*)wdtbf, dtm, b_dt,
        SEQ, D_INNER, DT_RANK, D_INNER);

    // sequential scan (all streams coalesced in (s,d))
    scan_kernel<<<D_INNER / 256, 256, 0, stream>>>(dtm, hsd, zsil, Bm, Cm, A_log, Dp, yraw);

    // rmsnorm over d -> bf16
    rmsnorm_scan_kernel<<<SEQ, 256, 0, stream>>>(yraw, scan_ln_w, ybf);

    // GEMM4: out[s,e] = sum_d y[s,d] * W_out[e,d]  (M=2048,N=2048,K=4096) -> d_out
    gemm_wmma_bf16<0><<<dim3(D_MODEL / 256, SEQ / 128), 256, 0, stream>>>(
        (const __bf16*)ybf, (const __bf16*)wobf, out, nullptr,
        SEQ, D_MODEL, D_INNER, D_MODEL);
}
